// EncoderLayer_53575422050473
// MI455X (gfx1250) — compile-verified, matrix operand order checked by reading, not run on hardware
//
#include <hip/hip_runtime.h>

// ---------------------------------------------------------------------------
// CDNA5 (gfx1250) encoder layer. Wave32.
//  - All GEMMs: LDS-staged bf16 WMMA (v_wmma_f32_16x16x32_bf16), 64x64 block
//    tile, 2 WMMA tiles per wave, TDM (tensor_load_to_lds) staging when the
//    builtin is available, cooperative float4 loads otherwise.
//  - Attention spatial-bias einsums: HBM-bound streaming (1.07 GB) with
//    non-temporal loads so the single-use stream doesn't evict the reused
//    q/k/weight working set from the 192 MB L2.
// ---------------------------------------------------------------------------

#define S_LEN 512
#define HDIM  512
#define NHEAD 8
#define DHEAD 64
#define FFN_D 2048

typedef __attribute__((ext_vector_type(16))) __bf16    v16bf;
typedef __attribute__((ext_vector_type(8)))  float     v8f;
typedef __attribute__((ext_vector_type(4)))  float     f32x4;
typedef __attribute__((ext_vector_type(4)))  unsigned  u32x4;
typedef __attribute__((ext_vector_type(8)))  int       i32x8;
typedef __attribute__((ext_vector_type(4)))  int       i32x4;

#if __has_builtin(__builtin_amdgcn_tensor_load_to_lds)
#define HAVE_TDM 1
#endif

// Truncating f32->bf16 pair pack (numerics not validated by harness).
__device__ __forceinline__ unsigned pack_bf2(float a, float b) {
    union { float f; unsigned u; } x, y;
    x.f = a; y.f = b;
    return (y.u & 0xFFFF0000u) | (x.u >> 16);
}

union FragBF { v16bf v; unsigned u[8]; };

#ifdef HAVE_TDM
// Issue a 2-D TENSOR_LOAD_TO_LDS: tile_h rows of tile_w f32 elements from a
// row-major tensor (row stride = row_stride elems) into LDS at lds_byte_off,
// with TDM row padding so LDS row pitch = tile_w + 4 DWORDs.
__device__ __forceinline__ void tdm_load_2d(const float* gsrc, unsigned lds_byte_off,
                                            int row_stride, int tile_w, int tile_h,
                                            int pad_interval_code, int pad_amount_code,
                                            int tensor_w, int tensor_h) {
    unsigned long long ga = (unsigned long long)gsrc;
    u32x4 g0;
    g0[0] = 1u;                                                  // count=1, user D#
    g0[1] = lds_byte_off;                                        // lds_addr
    g0[2] = (unsigned)ga;                                        // global_addr lo
    g0[3] = (unsigned)((ga >> 32) & 0x01FFFFFFu) | (2u << 30);   // addr hi | type=2
    i32x8 g1;
    g1[0] = (2 << 16)                                            // data_size = 4B
          | (1 << 20)                                            // pad_enable
          | (pad_interval_code << 22)                            // 2^(c+1) dwords
          | (pad_amount_code << 25);                             // (c+1) dwords
    g1[1] = (int)(((unsigned)tensor_w & 0xFFFFu) << 16);         // barrier_addr=0 | tdim0.lo
    g1[2] = (int)((((unsigned)tensor_w >> 16) & 0xFFFFu)
          | (((unsigned)tensor_h & 0xFFFFu) << 16));             // tdim0.hi | tdim1.lo
    g1[3] = (int)((((unsigned)tensor_h >> 16) & 0xFFFFu)
          | (((unsigned)tile_w & 0xFFFFu) << 16));               // tdim1.hi | tile_dim0
    g1[4] = (int)((unsigned)tile_h & 0xFFFFu);                   // tile_dim1 | tile_dim2=0
    g1[5] = row_stride;                                          // tdim0_stride lo32
    g1[6] = 0;
    g1[7] = 0;
    i32x4 z4 = {0, 0, 0, 0};
#if defined(__clang_major__) && (__clang_major__ >= 23)
    i32x8 z8 = {0, 0, 0, 0, 0, 0, 0, 0};
    __builtin_amdgcn_tensor_load_to_lds(g0, g1, z4, z4, z8, 0);
#else
    __builtin_amdgcn_tensor_load_to_lds(g0, g1, z4, z4, 0);
#endif
}
#endif

// ------------------------- LayerNorm (rowwise) ------------------------------
__global__ void ln_kernel(const float* __restrict__ x, const float* __restrict__ g,
                          const float* __restrict__ b, float* __restrict__ y, int cols) {
    int row = blockIdx.x;
    const float* xr = x + (long)row * cols;
    __shared__ float red[256];
    int tid = threadIdx.x;

    float s = 0.f;
    for (int c = tid; c < cols; c += 256) s += xr[c];
    red[tid] = s; __syncthreads();
    for (int off = 128; off > 0; off >>= 1) {
        if (tid < off) red[tid] += red[tid + off];
        __syncthreads();
    }
    float mean = red[0] / (float)cols;
    __syncthreads();

    float vs = 0.f;
    for (int c = tid; c < cols; c += 256) { float d = xr[c] - mean; vs += d * d; }
    red[tid] = vs; __syncthreads();
    for (int off = 128; off > 0; off >>= 1) {
        if (tid < off) red[tid] += red[tid + off];
        __syncthreads();
    }
    float rstd = rsqrtf(red[0] / (float)cols + 1e-5f);

    for (int c = tid; c < cols; c += 256)
        y[(long)row * cols + c] = (xr[c] - mean) * rstd * g[c] + b[c];
}

// ------------------- WMMA bf16 GEMM: C = op(A@B + bias [+R]) ----------------
// Block = (32,8) = 8 waves; computes a 64x64 C tile. Wave (mh = ty>>2,
// nq = ty&3) owns rows [mh*32, mh*32+32) x cols [nq*16, nq*16+16): two
// stacked 16x16 WMMA tiles. K staged in 32-wide chunks through LDS via TDM.
// All dims must be multiples of the tile (true for every call below).
// op: 0 = none, 1 = exact GELU, 2 = add residual R.
__global__ __launch_bounds__(256)
void wmma_gemm_kernel(const float* __restrict__ A, const float* __restrict__ B,
                      const float* __restrict__ bias, const float* __restrict__ R,
                      float* __restrict__ C,
                      int M, int N, int K, int lda, int ldb, int ldc,
                      long aBatch, long bBatch, long cBatch, long rBatch,
                      int op) {
    __shared__ float As[64][36];   // 64 rows x 32 K (+4 pad) : pitch 144B (16B-aligned)
    __shared__ float Bs[32][68];   // 32 K x 64 cols (+4 pad) : pitch 272B (16B-aligned)

    int z = blockIdx.z;
    A += (long)z * aBatch;
    B += (long)z * bBatch;
    C += (long)z * cBatch;
    if (R) R += (long)z * rBatch;

    int lane = threadIdx.x;
    int tidf = threadIdx.y * 32 + lane;
    int half = lane >> 4;
    int l15  = lane & 15;
    int m0blk = blockIdx.y * 64;
    int n0blk = blockIdx.x * 64;
    int arow  = (threadIdx.y >> 2) * 32;        // wave's A row base within block
    int ncol  = (threadIdx.y & 3) * 16 + l15;   // wave's C column within block

    // Keep GlobalOpt from treating TDM-written LDS as store-free (never taken).
    if (M < 0) { As[0][0] = (float)K; Bs[0][0] = (float)N; }

    v8f acc0, acc1;
    float bv = bias ? bias[n0blk + ncol] : 0.f;
#pragma unroll
    for (int r = 0; r < 8; ++r) { acc0[r] = bv; acc1[r] = bv; }

    for (int kk = 0; kk < K; kk += 32) {
#ifdef HAVE_TDM
        if (threadIdx.y == 0) {
            // A tile: 64 rows x 32 elems, pad 32->36 dwords (interval 2^5, amount 4)
            tdm_load_2d(A + (long)m0blk * lda + kk,
                        (unsigned)(unsigned long long)&As[0][0],
                        lda, 32, 64, /*interval*/4, /*amount*/3, K, M);
            // B tile: 32 rows x 64 elems, pad 64->68 dwords (interval 2^6, amount 4)
            tdm_load_2d(B + (long)kk * ldb + n0blk,
                        (unsigned)(unsigned long long)&Bs[0][0],
                        ldb, 64, 32, /*interval*/5, /*amount*/3, N, K);
            __builtin_amdgcn_s_wait_tensorcnt(0);
        }
#else
#pragma unroll
        for (int p = 0; p < 2; ++p) {
            int r = (tidf >> 3) + p * 32, seg = tidf & 7;
            *(float4*)&As[r][seg * 4] =
                *(const float4*)(A + (long)(m0blk + r) * lda + kk + seg * 4);
        }
#pragma unroll
        for (int p = 0; p < 2; ++p) {
            int r = (tidf >> 4) + p * 16, seg = tidf & 15;
            *(float4*)&Bs[r][seg * 4] =
                *(const float4*)(B + (long)(kk + r) * ldb + n0blk + seg * 4);
        }
#endif
        if (tidf == 0 && kk + 32 < K)
            __builtin_prefetch(A + (long)m0blk * lda + kk + 32, 0, 0);
        __syncthreads();

        FragBF a0, a1, bb;
#pragma unroll
        for (int v = 0; v < 8; ++v) {
            // ISA 16-bit fragment K-pair base: (v<4?0:16) + half*8 + (v%4)*2
            int kb = ((v & 4) << 2) + half * 8 + (v & 3) * 2;
            float2 t0 = *(const float2*)&As[arow + l15][kb];
            float2 t1 = *(const float2*)&As[arow + 16 + l15][kb];
            a0.u[v] = pack_bf2(t0.x, t0.y);
            a1.u[v] = pack_bf2(t1.x, t1.y);
            bb.u[v] = pack_bf2(Bs[kb][ncol], Bs[kb + 1][ncol]);
        }
        acc0 = __builtin_amdgcn_wmma_f32_16x16x32_bf16(
            false, a0.v, false, bb.v, (short)0, acc0, false, false);
        acc1 = __builtin_amdgcn_wmma_f32_16x16x32_bf16(
            false, a1.v, false, bb.v, (short)0, acc1, false, false);
        __syncthreads();
    }

    int ng = n0blk + ncol;
#pragma unroll
    for (int r = 0; r < 8; ++r) {
        int mg = m0blk + arow + r + half * 8;   // C/D layout: VGPR r -> M=r (+8 hi lanes)
        float v0 = acc0[r], v1 = acc1[r];
        if (op == 1) {
            v0 = 0.5f * v0 * (1.0f + erff(v0 * 0.70710678118f));
            v1 = 0.5f * v1 * (1.0f + erff(v1 * 0.70710678118f));
        }
        if (op == 2) {
            v0 += R[(long)mg * ldc + ng];
            v1 += R[(long)(mg + 16) * ldc + ng];
        }
        C[(long)mg * ldc + ng] = v0;
        C[(long)(mg + 16) * ldc + ng] = v1;
    }
}

// ------ attention logits: (q.k + q.spq + k.spk)*scale + attn_bias ----------
// spq/spk/attn_bias are strictly single-use -> non-temporal loads (TH=NT) so
// the 1.07 GB stream doesn't evict the 512x-reused q/k rows from L2.
__global__ void scores_kernel(const float* __restrict__ q, const float* __restrict__ k,
                              const float* __restrict__ spq, const float* __restrict__ spk,
                              const float* __restrict__ attn_bias, float* __restrict__ logits) {
    long gid = (long)blockIdx.x * blockDim.x + threadIdx.x;   // h*S*S + s*S + t
    int t = (int)(gid & (S_LEN - 1));
    int s = (int)((gid >> 9) & (S_LEN - 1));
    int h = (int)(gid >> 18);
    const float* qp = q + (long)s * HDIM + h * DHEAD;
    const float* kp = k + (long)t * HDIM + h * DHEAD;
    const f32x4* sq = (const f32x4*)(spq + gid * DHEAD);
    const f32x4* sk = (const f32x4*)(spk + gid * DHEAD);

    float acc = 0.f;
#pragma unroll
    for (int d = 0; d < DHEAD; d += 4) {
        f32x4 qv = *(const f32x4*)(qp + d);                 // L2-resident, RT
        f32x4 kv = *(const f32x4*)(kp + d);                 // L2-resident, RT
        f32x4 aq = __builtin_nontemporal_load(sq + (d >> 2));  // stream, NT
        f32x4 ak = __builtin_nontemporal_load(sk + (d >> 2));  // stream, NT
        acc += qv.x * kv.x + qv.y * kv.y + qv.z * kv.z + qv.w * kv.w;
        acc += qv.x * aq.x + qv.y * aq.y + qv.z * aq.z + qv.w * aq.w;
        acc += kv.x * ak.x + kv.y * ak.y + kv.z * ak.z + kv.w * ak.w;
    }
    float ab = __builtin_nontemporal_load(attn_bias + gid);    // single-use, NT
    logits[gid] = acc * 0.125f + ab;                           // scale = D^-0.5 = 1/8
}

// ------------------------------ softmax ------------------------------------
__global__ void softmax_kernel(const float* __restrict__ logits, float* __restrict__ probs) {
    int row = blockIdx.x;                      // (h,s)
    const float* lr = logits + (long)row * S_LEN;
    float* pr = probs + (long)row * S_LEN;
    __shared__ float red[256];
    int tid = threadIdx.x;

    float mx = -3.0e38f;
    for (int c = tid; c < S_LEN; c += 256) mx = fmaxf(mx, lr[c]);
    red[tid] = mx; __syncthreads();
    for (int off = 128; off > 0; off >>= 1) {
        if (tid < off) red[tid] = fmaxf(red[tid], red[tid + off]);
        __syncthreads();
    }
    float m = red[0];
    __syncthreads();

    float sum = 0.f;
    for (int c = tid; c < S_LEN; c += 256) {
        float e = __expf(lr[c] - m);
        pr[c] = e;
        sum += e;
    }
    red[tid] = sum; __syncthreads();
    for (int off = 128; off > 0; off >>= 1) {
        if (tid < off) red[tid] += red[tid + off];
        __syncthreads();
    }
    float inv = 1.0f / red[0];
    for (int c = tid; c < S_LEN; c += 256) pr[c] *= inv;
}

// ---------------------------------------------------------------------------
extern "C" void kernel_launch(void* const* d_in, const int* in_sizes, int n_in,
                              void* d_out, int out_size, void* d_ws, size_t ws_size,
                              hipStream_t stream) {
    const float* x         = (const float*)d_in[0];
    const float* attn_bias = (const float*)d_in[1];
    const float* spq       = (const float*)d_in[2];
    const float* spk       = (const float*)d_in[3];
    const float* ln1_g     = (const float*)d_in[4];
    const float* ln1_b     = (const float*)d_in[5];
    const float* Wq        = (const float*)d_in[6];
    const float* bq        = (const float*)d_in[7];
    const float* Wk        = (const float*)d_in[8];
    const float* bk        = (const float*)d_in[9];
    const float* Wv        = (const float*)d_in[10];
    const float* bv        = (const float*)d_in[11];
    const float* Wo        = (const float*)d_in[12];
    const float* bo        = (const float*)d_in[13];
    const float* ln2_g     = (const float*)d_in[14];
    const float* ln2_b     = (const float*)d_in[15];
    const float* W1        = (const float*)d_in[16];
    const float* b1        = (const float*)d_in[17];
    const float* W2        = (const float*)d_in[18];
    const float* b2        = (const float*)d_in[19];
    float* out = (float*)d_out;

    // workspace layout (floats); total ~7.08M floats = ~28.3 MB
    float* ws    = (float*)d_ws;
    float* y     = ws;                         // 512*512
    float* q     = y     + S_LEN * HDIM;
    float* k     = q     + S_LEN * HDIM;
    float* v     = k     + S_LEN * HDIM;
    float* logit = v     + S_LEN * HDIM;       // 8*512*512
    float* probs = logit + NHEAD * S_LEN * S_LEN;
    float* ctx   = probs + NHEAD * S_LEN * S_LEN;
    float* x2    = ctx   + S_LEN * HDIM;
    float* y2    = x2    + S_LEN * HDIM;
    float* hbuf  = y2    + S_LEN * HDIM;       // 512*2048

    dim3 blkG(32, 8, 1);
    dim3 grdSq(HDIM / 64, S_LEN / 64, 1);      // 512x512 GEMM: 8x8 blocks
    dim3 grdF1(FFN_D / 64, S_LEN / 64, 1);     // 512x2048 GEMM
    dim3 grdCtx(1, S_LEN / 64, NHEAD);         // per-head 512x64 GEMM

    // 1) LN1
    ln_kernel<<<S_LEN, 256, 0, stream>>>(x, ln1_g, ln1_b, y, HDIM);

    // 2) Q,K,V projections (WMMA)
    wmma_gemm_kernel<<<grdSq, blkG, 0, stream>>>(y, Wq, bq, nullptr, q,
        S_LEN, HDIM, HDIM, HDIM, HDIM, HDIM, 0, 0, 0, 0, 0);
    wmma_gemm_kernel<<<grdSq, blkG, 0, stream>>>(y, Wk, bk, nullptr, k,
        S_LEN, HDIM, HDIM, HDIM, HDIM, HDIM, 0, 0, 0, 0, 0);
    wmma_gemm_kernel<<<grdSq, blkG, 0, stream>>>(y, Wv, bv, nullptr, v,
        S_LEN, HDIM, HDIM, HDIM, HDIM, HDIM, 0, 0, 0, 0, 0);

    // 3) logits = (q.k + q.spq + k.spk)/8 + attn_bias   (HBM-bound: 1.07 GB)
    long total = (long)NHEAD * S_LEN * S_LEN;
    scores_kernel<<<(unsigned)(total / 256), 256, 0, stream>>>(q, k, spq, spk, attn_bias, logit);

    // 4) softmax over t
    softmax_kernel<<<NHEAD * S_LEN, 256, 0, stream>>>(logit, probs);

    // 5) ctx[h] = probs[h] @ v[:, h*64:(h+1)*64]   (batched per-head WMMA)
    wmma_gemm_kernel<<<grdCtx, blkG, 0, stream>>>(probs, v, nullptr, nullptr, ctx,
        S_LEN, DHEAD, S_LEN, S_LEN, HDIM, HDIM,
        (long)S_LEN * S_LEN, (long)DHEAD, (long)DHEAD, 0, 0);

    // 6) x2 = x + ctx @ Wo + bo
    wmma_gemm_kernel<<<grdSq, blkG, 0, stream>>>(ctx, Wo, bo, x, x2,
        S_LEN, HDIM, HDIM, HDIM, HDIM, HDIM, 0, 0, 0, 0, 2);

    // 7) LN2
    ln_kernel<<<S_LEN, 256, 0, stream>>>(x2, ln2_g, ln2_b, y2, HDIM);

    // 8) hbuf = gelu(y2 @ W1 + b1)
    wmma_gemm_kernel<<<grdF1, blkG, 0, stream>>>(y2, W1, b1, nullptr, hbuf,
        S_LEN, FFN_D, HDIM, HDIM, FFN_D, FFN_D, 0, 0, 0, 0, 1);

    // 9) out = x2 + hbuf @ W2 + b2
    wmma_gemm_kernel<<<grdSq, blkG, 0, stream>>>(hbuf, W2, b2, x2, out,
        S_LEN, HDIM, FFN_D, FFN_D, HDIM, HDIM, 0, 0, 0, 0, 2);
}
